// SelfAttention_7559142441246
// MI455X (gfx1250) — matrix operand order repaired and checked by reading, not compiled, not run on hardware
//
#include <hip/hip_runtime.h>
#include <hip/hip_bf16.h>

// Self-attention (b=4, c=256, n=4096) for MI455X gfx1250.
// All four matmuls run on v_wmma_f32_16x16x32_bf16 (wave32). Flash-style
// online softmax keeps the 4096x4096 score matrices out of HBM.
// Contiguous global->LDS staging uses GLOBAL_LOAD_ASYNC_TO_LDS (ASYNCcnt).
// Workspace layout (bytes):
//   xb    @ 0          : 8,388,608  (x in bf16)
//   wqkvb @ 8,388,608  :   393,216
//   woutb @ 8,781,824  :   131,072
//   qb    @ 8,912,896  : 8,388,608  (q, pre-scaled by 1/16)
//   kb    @ 17,301,504 : 8,388,608
//   vb    @ 25,690,112 : 8,388,608
//   ob    @ 34,078,720 : 8,388,608  (attention output, bf16)
// total ~42.5 MB.

typedef __attribute__((ext_vector_type(16))) __bf16 v16bf;
typedef __attribute__((ext_vector_type(8)))  float  v8f;
typedef __attribute__((ext_vector_type(8)))  unsigned int v8u;
typedef __attribute__((ext_vector_type(2)))  int    v2i;

#define GLB_AS __attribute__((address_space(1)))
#define LDS_AS __attribute__((address_space(3)))

#if __has_builtin(__builtin_amdgcn_global_load_async_to_lds_b32) && \
    __has_builtin(__builtin_amdgcn_global_load_async_to_lds_b64) && \
    __has_builtin(__builtin_amdgcn_s_wait_asynccnt)
#define USE_ASYNC 1
#else
#define USE_ASYNC 0
#endif

__device__ __forceinline__ void cp_b32(void* lds, const void* g) {
#if USE_ASYNC
  __builtin_amdgcn_global_load_async_to_lds_b32(
      (GLB_AS int*)g, (LDS_AS int*)lds, 0, 0);
#else
  *(unsigned int*)lds = *(const unsigned int*)g;
#endif
}
__device__ __forceinline__ void cp_b64(void* lds, const void* g) {
#if USE_ASYNC
  __builtin_amdgcn_global_load_async_to_lds_b64(
      (GLB_AS v2i*)g, (LDS_AS v2i*)lds, 0, 0);
#else
  const unsigned int* s = (const unsigned int*)g;
  unsigned int* d = (unsigned int*)lds;
  unsigned int a = s[0], b = s[1];
  d[0] = a; d[1] = b;
#endif
}
__device__ __forceinline__ void wait_async() {
#if USE_ASYNC
  __builtin_amdgcn_s_wait_asynccnt(0);
#endif
}

__device__ __forceinline__ unsigned short f2bf(float f) {
  unsigned int u = __float_as_uint(f);
  u += 0x7FFFu + ((u >> 16) & 1u);           // round-to-nearest-even
  return (unsigned short)(u >> 16);
}
__device__ __forceinline__ unsigned int pack2(unsigned short lo, unsigned short hi) {
  return (unsigned int)lo | ((unsigned int)hi << 16);
}
__device__ __forceinline__ v16bf frag_from(const unsigned int* w) {
  v8u t;
  t[0]=w[0]; t[1]=w[1]; t[2]=w[2]; t[3]=w[3];
  t[4]=w[4]; t[5]=w[5]; t[6]=w[6]; t[7]=w[7];
  return __builtin_bit_cast(v16bf, t);
}

// ---------------------------------------------------------------- convert
__global__ __launch_bounds__(256) void cvt_bf16(const float* __restrict__ src,
                                                unsigned short* __restrict__ dst,
                                                int n) {
  int i = blockIdx.x * 256 + threadIdx.x;
  int stride = gridDim.x * 256;
  for (; i < n; i += stride) dst[i] = f2bf(src[i]);
}

// ---------------------------------------------------------------- QKV GEMM
// QKV[768,4096] = W[768,256] @ X[256,4096] + bias, per batch.
// Block tile 128x128, 8 waves (2x4), wave tile 64x32, K-step 32.
__global__ __launch_bounds__(256) void qkv_gemm(
    const unsigned short* __restrict__ W,     // [768,256] bf16 row-major
    const unsigned short* __restrict__ X,     // [B,256,4096] bf16
    const float* __restrict__ bias,           // [768]
    unsigned short* __restrict__ Qo,          // [B,256,4096]
    unsigned short* __restrict__ Ko,
    unsigned short* __restrict__ Vo) {
  __shared__ unsigned short At[128 * 36];     // A tile [m][k], pitch 36
  __shared__ unsigned int   Bp[16 * 128];     // B tile, K-pair interleaved [k2][n]
  const int tid  = threadIdx.x;
  const int lane = tid & 31;
  const int wave = tid >> 5;
  const int wm   = wave >> 2;                 // 0..1
  const int wn   = wave & 3;                  // 0..3
  const int hf   = lane >> 4;
  const int l16  = lane & 15;
  const int bM   = blockIdx.y * 128;
  const int bN   = blockIdx.x * 128;
  const int b    = blockIdx.z;
  const unsigned short* Xb = X + (size_t)b * 256 * 4096;

  // uniform per block: rows bM..bM+127 live entirely inside one of q/k/v
  const int which = blockIdx.y >> 1;          // 0=q, 1=k, 2=v (scalar)
  unsigned short* __restrict__ dst = (which == 0) ? Qo : (which == 1) ? Ko : Vo;
  const float scale = (which == 0) ? 0.0625f : 1.0f;   // fold 1/sqrt(c) into q

  v8f acc[4][2] = {};

  for (int kk = 0; kk < 8; ++kk) {
    const int k0 = kk * 32;
    __syncthreads();
    // stage A (contiguous 8B chunks) via async global->LDS DMA
    #pragma unroll
    for (int i = 0; i < 4; ++i) {
      int idx = tid + 256 * i;                // 0..1023
      int m   = idx >> 3;
      int kq  = (idx & 7) * 4;
      cp_b64(At + m * 36 + kq, W + (size_t)(bM + m) * 256 + k0 + kq);
    }
    // stage B: pair-interleave K so B fragments are single b32 LDS loads
    #pragma unroll
    for (int i = 0; i < 8; ++i) {
      int idx = tid + 256 * i;                // 0..2047
      int k2  = idx >> 7;                     // 0..15
      int n   = idx & 127;
      int c   = k0 + k2 * 2;
      unsigned short g0 = Xb[(size_t)c * 4096 + bN + n];
      unsigned short g1 = Xb[(size_t)(c + 1) * 4096 + bN + n];
      Bp[k2 * 128 + n] = pack2(g0, g1);
    }
    wait_async();
    __syncthreads();

    v16bf af[4], bfr[2];
    #pragma unroll
    for (int mt = 0; mt < 4; ++mt) {
      int row = wm * 64 + mt * 16 + l16;
      const unsigned short* base = At + row * 36 + hf * 8;
      unsigned int w[8];
      #pragma unroll
      for (int v = 0; v < 4; ++v) w[v]     = *(const unsigned int*)(base + v * 2);
      #pragma unroll
      for (int v = 0; v < 4; ++v) w[4 + v] = *(const unsigned int*)(base + 16 + v * 2);
      af[mt] = frag_from(w);
    }
    #pragma unroll
    for (int nt = 0; nt < 2; ++nt) {
      int col = wn * 32 + nt * 16 + l16;
      unsigned int w[8];
      #pragma unroll
      for (int v = 0; v < 8; ++v) w[v] = Bp[(hf * 4 + v + (v & 4)) * 128 + col];
      bfr[nt] = frag_from(w);
    }
    #pragma unroll
    for (int mt = 0; mt < 4; ++mt)
      #pragma unroll
      for (int nt = 0; nt < 2; ++nt)
        acc[mt][nt] = __builtin_amdgcn_wmma_f32_16x16x32_bf16(
            false, af[mt], false, bfr[nt], (short)0, acc[mt][nt], false, false);
  }

  #pragma unroll
  for (int mt = 0; mt < 4; ++mt) {
    #pragma unroll
    for (int nt = 0; nt < 2; ++nt) {
      int gn = bN + wn * 32 + nt * 16 + l16;
      #pragma unroll
      for (int r = 0; r < 8; ++r) {
        int orow = bM + wm * 64 + mt * 16 + r + 8 * hf;
        float val = (acc[mt][nt][r] + bias[orow]) * scale;
        size_t off = ((size_t)b * 256 + (orow & 255)) * 4096 + gn;
        dst[off] = f2bf(val);
      }
    }
  }
}

// ---------------------------------------------------------------- flash attention
// Per WG: one batch, 64 query columns. 4 waves, 16 columns each.
// S^T[n,m] = sum_c k[c,n]*q[c,m]  (q pre-scaled), online softmax over n,
// O^T[c,m] += v[c,n]*P[n,m]. 32 keys per iteration, 128 iterations.
__global__ __launch_bounds__(128) void flash_attn(
    const unsigned short* __restrict__ Q,     // [B,256,4096] bf16, scaled
    const unsigned short* __restrict__ K,
    const unsigned short* __restrict__ V,
    unsigned short* __restrict__ O) {         // [B,256,4096] bf16
  __shared__ unsigned int   Qp[128 * 64];     // q pairs [c2][m], staged once (32KB)
  __shared__ unsigned short KT[32 * 258];     // k transposed [n][c], pitch 258
  __shared__ unsigned short VT[256 * 34];     // v natural [c][n], pitch 34
  __shared__ unsigned int   Pp[4][16 * 16];   // per-wave P pairs [k2][m]
  const int tid  = threadIdx.x;
  const int lane = tid & 31;
  const int wave = tid >> 5;                  // 0..3
  const int hf   = lane >> 4;
  const int l16  = lane & 15;
  const int b    = blockIdx.y;
  const int m0   = blockIdx.x * 64;
  const unsigned short* Qb = Q + (size_t)b * 256 * 4096;
  const unsigned short* Kb = K + (size_t)b * 256 * 4096;
  const unsigned short* Vb = V + (size_t)b * 256 * 4096;

  // stage Q block once: pairs over c (128 pair-rows x 64 cols)
  for (int i = 0; i < 64; ++i) {
    int idx = tid + 128 * i;
    int c2  = idx >> 6;
    int m   = idx & 63;
    unsigned short g0 = Qb[(size_t)(2 * c2)     * 4096 + m0 + m];
    unsigned short g1 = Qb[(size_t)(2 * c2 + 1) * 4096 + m0 + m];
    Qp[c2 * 64 + m] = pack2(g0, g1);
  }

  v8f oacc[16] = {};
  float mrun = -3.0e38f;
  float rrun = 0.0f;
  const int mcol = wave * 16 + l16;           // column within the 64-block

  for (int n0 = 0; n0 < 4096; n0 += 32) {
    __syncthreads();                          // protect KT/VT reads of prev iter
    {
      // V tile (natural [c][n] layout): async global->LDS b32 copies
      int n2 = (tid & 15) * 2;                // even n
      int cv = tid >> 4;                      // 0..7
      for (int c = cv; c < 256; c += 8)
        cp_b32(&VT[c * 34 + n2], Vb + (size_t)c * 4096 + n0 + n2);
      // K tile transposed [n][c]: manual (non-contiguous LDS dst)
      int n  = tid & 31;
      int c0 = tid >> 5;                      // 0..3
      for (int c = c0; c < 256; c += 4)
        KT[n * 258 + c] = Kb[(size_t)c * 4096 + n0 + n];
      // prefetch next block's k/v rows so HBM overlaps this block's wmma
      if (n0 + 32 < 4096) {
        const unsigned short* kp = Kb + (size_t)(2 * tid) * 4096 + n0 + 32;
        const unsigned short* vp = Vb + (size_t)(2 * tid) * 4096 + n0 + 32;
        __builtin_prefetch(kp, 0, 1);
        __builtin_prefetch(kp + 4096, 0, 1);
        __builtin_prefetch(vp, 0, 1);
        __builtin_prefetch(vp + 4096, 0, 1);
      }
      wait_async();
    }
    __syncthreads();

    // S^T block: 2 n-tiles x 16 m, K-loop over c (8 x wmma each)
    v8f s[2] = {};
    #pragma unroll
    for (int kk = 0; kk < 8; ++kk) {
      unsigned int wq[8];
      #pragma unroll
      for (int v = 0; v < 8; ++v)
        wq[v] = Qp[(kk * 16 + hf * 4 + v + (v & 4)) * 64 + mcol];
      v16bf qf = frag_from(wq);
      #pragma unroll
      for (int t = 0; t < 2; ++t) {
        const unsigned short* base = KT + (t * 16 + l16) * 258 + kk * 32 + hf * 8;
        unsigned int w[8];
        #pragma unroll
        for (int v = 0; v < 4; ++v) w[v]     = *(const unsigned int*)(base + v * 2);
        #pragma unroll
        for (int v = 0; v < 4; ++v) w[4 + v] = *(const unsigned int*)(base + 16 + v * 2);
        v16bf kf = frag_from(w);
        s[t] = __builtin_amdgcn_wmma_f32_16x16x32_bf16(
            false, kf, false, qf, (short)0, s[t], false, false);
      }
    }

    // online softmax over the 32 n-rows of this block (per column m);
    // each column lives in lanes L and L+16 -> combine via shfl_xor 16.
    float lmax = -3.0e38f;
    #pragma unroll
    for (int t = 0; t < 2; ++t)
      #pragma unroll
      for (int r = 0; r < 8; ++r) lmax = fmaxf(lmax, s[t][r]);
    lmax = fmaxf(lmax, __shfl_xor(lmax, 16, 32));
    float mnew  = fmaxf(mrun, lmax);
    float alpha = __expf(mrun - mnew);
    float p[2][8];
    float lsum = 0.0f;
    #pragma unroll
    for (int t = 0; t < 2; ++t)
      #pragma unroll
      for (int r = 0; r < 8; ++r) {
        float e = __expf(s[t][r] - mnew);
        p[t][r] = e;
        lsum += e;
      }
    lsum += __shfl_xor(lsum, 16, 32);
    rrun = rrun * alpha + lsum;
    mrun = mnew;
    #pragma unroll
    for (int ct = 0; ct < 16; ++ct)
      #pragma unroll
      for (int r = 0; r < 8; ++r) oacc[ct][r] *= alpha;

    // write P as bf16 K-pairs into this wave's B-fragment-layout region
    unsigned int* pp = Pp[wave];
    #pragma unroll
    for (int t = 0; t < 2; ++t) {
      #pragma unroll
      for (int rp = 0; rp < 4; ++rp) {
        int nloc = t * 16 + hf * 8 + rp * 2;
        pp[(nloc >> 1) * 16 + l16] = pack2(f2bf(p[t][rp * 2]), f2bf(p[t][rp * 2 + 1]));
      }
    }
    __syncthreads();                          // order Pp write -> read

    // O^T += V * P : one wmma (K=32) per 16-row c tile
    unsigned int wp[8];
    #pragma unroll
    for (int v = 0; v < 8; ++v) wp[v] = pp[(hf * 4 + v + (v & 4)) * 16 + l16];
    v16bf pf = frag_from(wp);
    #pragma unroll
    for (int ct = 0; ct < 16; ++ct) {
      const unsigned short* base = VT + (ct * 16 + l16) * 34 + hf * 8;
      unsigned int w[8];
      #pragma unroll
      for (int v = 0; v < 4; ++v) w[v]     = *(const unsigned int*)(base + v * 2);
      #pragma unroll
      for (int v = 0; v < 4; ++v) w[4 + v] = *(const unsigned int*)(base + 16 + v * 2);
      v16bf vf = frag_from(w);
      oacc[ct] = __builtin_amdgcn_wmma_f32_16x16x32_bf16(
          false, vf, false, pf, (short)0, oacc[ct], false, false);
    }
  }

  float inv = 1.0f / rrun;
  #pragma unroll
  for (int ct = 0; ct < 16; ++ct) {
    #pragma unroll
    for (int r = 0; r < 8; ++r) {
      int c = ct * 16 + r + 8 * hf;
      O[((size_t)b * 256 + c) * 4096 + m0 + mcol] = f2bf(oacc[ct][r] * inv);
    }
  }
}

// ---------------------------------------------------------------- output GEMM
// out[256,4096] = Wout[256,256] @ O[256,4096] + bias, f32 result.
__global__ __launch_bounds__(256) void out_gemm(
    const unsigned short* __restrict__ W,     // [256,256] bf16
    const unsigned short* __restrict__ X,     // [B,256,4096] bf16 (attn output)
    const float* __restrict__ bias,           // [256]
    float* __restrict__ Out) {                // [B,256,4096] f32
  __shared__ unsigned short At[128 * 36];
  __shared__ unsigned int   Bp[16 * 128];
  const int tid  = threadIdx.x;
  const int lane = tid & 31;
  const int wave = tid >> 5;
  const int wm   = wave >> 2;
  const int wn   = wave & 3;
  const int hf   = lane >> 4;
  const int l16  = lane & 15;
  const int bM   = blockIdx.y * 128;
  const int bN   = blockIdx.x * 128;
  const int b    = blockIdx.z;
  const unsigned short* Xb = X + (size_t)b * 256 * 4096;

  v8f acc[4][2] = {};

  for (int kk = 0; kk < 8; ++kk) {
    const int k0 = kk * 32;
    __syncthreads();
    #pragma unroll
    for (int i = 0; i < 4; ++i) {
      int idx = tid + 256 * i;
      int m   = idx >> 3;
      int kq  = (idx & 7) * 4;
      cp_b64(At + m * 36 + kq, W + (size_t)(bM + m) * 256 + k0 + kq);
    }
    #pragma unroll
    for (int i = 0; i < 8; ++i) {
      int idx = tid + 256 * i;
      int k2  = idx >> 7;
      int n   = idx & 127;
      int c   = k0 + k2 * 2;
      unsigned short g0 = Xb[(size_t)c * 4096 + bN + n];
      unsigned short g1 = Xb[(size_t)(c + 1) * 4096 + bN + n];
      Bp[k2 * 128 + n] = pack2(g0, g1);
    }
    wait_async();
    __syncthreads();

    v16bf af[4], bfr[2];
    #pragma unroll
    for (int mt = 0; mt < 4; ++mt) {
      int row = wm * 64 + mt * 16 + l16;
      const unsigned short* base = At + row * 36 + hf * 8;
      unsigned int w[8];
      #pragma unroll
      for (int v = 0; v < 4; ++v) w[v]     = *(const unsigned int*)(base + v * 2);
      #pragma unroll
      for (int v = 0; v < 4; ++v) w[4 + v] = *(const unsigned int*)(base + 16 + v * 2);
      af[mt] = frag_from(w);
    }
    #pragma unroll
    for (int nt = 0; nt < 2; ++nt) {
      int col = wn * 32 + nt * 16 + l16;
      unsigned int w[8];
      #pragma unroll
      for (int v = 0; v < 8; ++v) w[v] = Bp[(hf * 4 + v + (v & 4)) * 128 + col];
      bfr[nt] = frag_from(w);
    }
    #pragma unroll
    for (int mt = 0; mt < 4; ++mt)
      #pragma unroll
      for (int nt = 0; nt < 2; ++nt)
        acc[mt][nt] = __builtin_amdgcn_wmma_f32_16x16x32_bf16(
            false, af[mt], false, bfr[nt], (short)0, acc[mt][nt], false, false);
  }

  #pragma unroll
  for (int mt = 0; mt < 4; ++mt) {
    #pragma unroll
    for (int nt = 0; nt < 2; ++nt) {
      int gn = bN + wn * 32 + nt * 16 + l16;
      #pragma unroll
      for (int r = 0; r < 8; ++r) {
        int orow = bM + wm * 64 + mt * 16 + r + 8 * hf;
        Out[((size_t)b * 256 + orow) * 4096 + gn] = acc[mt][nt][r] + bias[orow];
      }
    }
  }
}

// ---------------------------------------------------------------- launch
extern "C" void kernel_launch(void* const* d_in, const int* in_sizes, int n_in,
                              void* d_out, int out_size, void* d_ws, size_t ws_size,
                              hipStream_t stream) {
  const float* x     = (const float*)d_in[0];
  const float* qkv_w = (const float*)d_in[1];
  const float* qkv_b = (const float*)d_in[2];
  const float* out_w = (const float*)d_in[3];
  const float* out_b = (const float*)d_in[4];
  float* out = (float*)d_out;

  char* ws = (char*)d_ws;
  unsigned short* xb    = (unsigned short*)(ws);
  unsigned short* wqkvb = (unsigned short*)(ws + 8388608);
  unsigned short* woutb = (unsigned short*)(ws + 8781824);
  unsigned short* qb    = (unsigned short*)(ws + 8912896);
  unsigned short* kb    = (unsigned short*)(ws + 17301504);
  unsigned short* vb    = (unsigned short*)(ws + 25690112);
  unsigned short* ob    = (unsigned short*)(ws + 34078720);

  cvt_bf16<<<2048, 256, 0, stream>>>(x, xb, 4 * 256 * 4096);
  cvt_bf16<<<256, 256, 0, stream>>>(qkv_w, wqkvb, 768 * 256);
  cvt_bf16<<<64, 256, 0, stream>>>(out_w, woutb, 256 * 256);

  qkv_gemm<<<dim3(32, 6, 4), 256, 0, stream>>>(wqkvb, xb, qkv_b, qb, kb, vb);
  flash_attn<<<dim3(64, 4), 128, 0, stream>>>(qb, kb, vb, ob);
  out_gemm<<<dim3(32, 2, 4), 256, 0, stream>>>(woutb, ob, out_b, out);
}
